// MemristorLinear_5961414607275
// MI455X (gfx1250) — compile-verified, matrix operand order checked by reading, not run on hardware
//
#include <hip/hip_runtime.h>
#include <stdint.h>

typedef __attribute__((ext_vector_type(16))) __bf16 v16bf;
typedef __attribute__((ext_vector_type(8)))  float  v8f;

#define B_DIM   2048
#define IN_DIM  4096
#define OUT_DIM 4096
#define KTILES  32          // IN / 128 ADC tiles
#define BM      128
#define BN      128
#define NBLK    ((B_DIM / BM) * (OUT_DIM / BN))   // 16 * 32 = 512

// physical constants
#define GMIN_C   1e-5f
#define GMAX_C   1e-4f
#define GRANGE_C 9e-5f
#define D2D_C    (0.01f * GRANGE_C)
#define RN_C     (0.02f * GRANGE_C)
#define OUT_SCALE (1.0f / GRANGE_C)   // W_MAX / G_RANGE

// workspace layout (bytes)
//   ctl[0]        : xmax bits
//   ctl[32 + t]   : per-tile ADC max bits
//   ctl[64 + t]   : per-tile arrival counter
#define CTL_WORDS 256
#define XQ_OFF 4096
#define G_OFF  (XQ_OFF + (size_t)B_DIM * IN_DIM * 2)

__device__ __forceinline__ unsigned short f32_to_bf16_rne(float f) {
    unsigned u = __float_as_uint(f);
    unsigned r = u + 0x7fffu + ((u >> 16) & 1u);
    return (unsigned short)(r >> 16);
}

__global__ void mx_init_kernel(unsigned* ctl) {
    if (threadIdx.x < CTL_WORDS) ctl[threadIdx.x] = 0u;
}

// ---- global max|x| reduction (float4 loads, wave32 shuffle, uint atomicMax) ----
__global__ void mx_absmax_kernel(const float* __restrict__ x, unsigned* __restrict__ ctl, int n4) {
    const float4* xv = (const float4*)x;
    float m = 0.f;
    for (int i = blockIdx.x * blockDim.x + threadIdx.x; i < n4; i += gridDim.x * blockDim.x) {
        float4 v = xv[i];
        m = fmaxf(m, fmaxf(fmaxf(fabsf(v.x), fabsf(v.y)), fmaxf(fabsf(v.z), fabsf(v.w))));
    }
    for (int s = 16; s; s >>= 1) m = fmaxf(m, __shfl_xor(m, s, 32));
    __shared__ float sm[8];
    int lane = threadIdx.x & 31, w = threadIdx.x >> 5;
    if (!lane) sm[w] = m;
    __syncthreads();
    if (threadIdx.x == 0) {
        float bm = sm[0];
        for (int i = 1; i < 8; i++) bm = fmaxf(bm, sm[i]);
        atomicMax(&ctl[0], __float_as_uint(bm));   // values are >= 0: uint order == float order
    }
}

// ---- DAC quantize x -> bf16 ----
__global__ void mx_quantx_kernel(const float* __restrict__ x, const unsigned* __restrict__ ctl,
                                 unsigned short* __restrict__ xq, int n4) {
    float mx   = __uint_as_float(ctl[0]);
    float step = mx > 0.f ? mx * (1.f / 127.f) : 1.f;
    float inv  = 1.f / step;
    const float4* xv = (const float4*)x;
    ushort4* qv = (ushort4*)xq;
    for (int i = blockIdx.x * blockDim.x + threadIdx.x; i < n4; i += gridDim.x * blockDim.x) {
        float4 v = xv[i];
        float a = fminf(fmaxf(rintf(v.x * inv), -127.f), 127.f) * step;
        float b = fminf(fmaxf(rintf(v.y * inv), -127.f), 127.f) * step;
        float c = fminf(fmaxf(rintf(v.z * inv), -127.f), 127.f) * step;
        float d = fminf(fmaxf(rintf(v.w * inv), -127.f), 127.f) * step;
        ushort4 o;
        o.x = f32_to_bf16_rne(a); o.y = f32_to_bf16_rne(b);
        o.z = f32_to_bf16_rne(c); o.w = f32_to_bf16_rne(d);
        qv[i] = o;
    }
}

// ---- build effective conductance matrix -> bf16 ----
__global__ void mx_geff_kernel(const float* __restrict__ W, const float* __restrict__ dp,
                               const float* __restrict__ dn, const float* __restrict__ np_,
                               const float* __restrict__ nn, unsigned short* __restrict__ g, int n4) {
    const float4* Wv  = (const float4*)W;
    const float4* dpv = (const float4*)dp;
    const float4* dnv = (const float4*)dn;
    const float4* npv = (const float4*)np_;
    const float4* nnv = (const float4*)nn;
    ushort4* gv = (ushort4*)g;
    for (int i = blockIdx.x * blockDim.x + threadIdx.x; i < n4; i += gridDim.x * blockDim.x) {
        float4 w = Wv[i], a = dpv[i], b = dnv[i], c = npv[i], d = nnv[i];
        float wf[4] = {w.x, w.y, w.z, w.w};
        float af[4] = {a.x, a.y, a.z, a.w};
        float bf[4] = {b.x, b.y, b.z, b.w};
        float cf[4] = {c.x, c.y, c.z, c.w};
        float df[4] = {d.x, d.y, d.z, d.w};
        unsigned short of[4];
        #pragma unroll
        for (int e = 0; e < 4; e++) {
            float wc = fminf(fmaxf(wf[e], -1.f), 1.f);
            float gp = GMIN_C + fmaxf(wc, 0.f) * GRANGE_C;
            float gn = GMIN_C + fmaxf(-wc, 0.f) * GRANGE_C;
            float Gp = fminf(fmaxf(gp + D2D_C * af[e] + RN_C * cf[e], 0.f), GMAX_C);
            float Gn = fminf(fmaxf(gn + D2D_C * bf[e] + RN_C * df[e], 0.f), GMAX_C);
            of[e] = f32_to_bf16_rne(Gp - Gn);
        }
        ushort4 o; o.x = of[0]; o.y = of[1]; o.z = of[2]; o.w = of[3];
        gv[i] = o;
    }
}

// ---- persistent fused GEMM: double-buffered async global->LDS staging,
//      32 ADC tiles, grid-wide max per tile, WMMA bf16 ----
__global__ __launch_bounds__(256, 1) void mx_gemm_kernel(
        const unsigned short* __restrict__ xq,   // [B][IN]  bf16 bits
        const unsigned short* __restrict__ g,    // [OUT][IN] bf16 bits
        const float* __restrict__ bias,
        unsigned* __restrict__ ctl,
        float* __restrict__ out) {
    // padded rows: stride 136 halfwords (272B = 17*16B) -> conflict-free ds_load_b128
    __shared__ unsigned short lA[2][BM][136];
    __shared__ unsigned short lB[2][BN][136];
    __shared__ float sMax[8];

    const int tid  = threadIdx.x;
    const int lane = tid & 31;
    const int wave = tid >> 5;
    const int bm   = blockIdx.x & 15;   // 16 M-tiles
    const int bn   = blockIdx.x >> 4;   // 32 N-tiles
    const int r    = lane & 15;
    const int hi   = lane >> 4;

    // uniform 64-bit global bases (SGPRs) for GVS-mode async loads
    const unsigned long long baseA =
        (unsigned long long)(uintptr_t)(xq + (size_t)(bm * BM) * IN_DIM);
    const unsigned long long baseB =
        (unsigned long long)(uintptr_t)(g + (size_t)(bn * BN) * IN_DIM);

    // issue this thread's 16 async b128 copies (8 for A, 8 for B) of tile t into buf
    auto issue_tile = [&](int t, int buf) {
        // low 32 bits of a flat shared pointer == wave-relative LDS byte offset
        unsigned ldsA0 = (unsigned)(uintptr_t)&lA[buf][0][0];
        unsigned ldsB0 = (unsigned)(uintptr_t)&lB[buf][0][0];
        #pragma unroll
        for (int i = 0; i < 8; i++) {
            int idx = i * 256 + tid;
            int row = idx >> 4;                 // 16 x uint4 per 128-halfword row slice
            int c   = idx & 15;
            unsigned voff = (unsigned)(row * (IN_DIM * 2) + t * 256 + c * 16); // bytes
            unsigned loff = (unsigned)(row * 272 + c * 16);                    // padded LDS row
            unsigned la = ldsA0 + loff;
            unsigned lb = ldsB0 + loff;
            asm volatile("global_load_async_to_lds_b128 %0, %1, %2"
                         :: "v"(la), "v"(voff), "s"(baseA) : "memory");
            asm volatile("global_load_async_to_lds_b128 %0, %1, %2"
                         :: "v"(lb), "v"(voff), "s"(baseB) : "memory");
        }
    };

    const v8f vzero = {0.f, 0.f, 0.f, 0.f, 0.f, 0.f, 0.f, 0.f};
    v8f tot[8], acc[8];
    #pragma unroll
    for (int j = 0; j < 8; j++) { tot[j] = vzero; acc[j] = vzero; }

    issue_tile(0, 0);   // prologue: prefetch tile 0

    for (int t = 0; t < KTILES; ++t) {
        const int buf = t & 1;
        // prefetch tile t+1 into the other buffer, then wait only for tile t's
        // 16 in-order async loads (ASYNCcnt <= 16 leaves t+1's in flight)
        if (t + 1 < KTILES) {
            issue_tile(t + 1, buf ^ 1);
            asm volatile("s_wait_asynccnt 0x10" ::: "memory");
        } else {
            asm volatile("s_wait_asynccnt 0x0" ::: "memory");
        }
        __syncthreads();

        // ---- 4 WMMA K-steps of 32 cover this 128-wide ADC tile ----
        #pragma unroll
        for (int kk = 0; kk < 4; kk++) {
            union { uint4 u[2]; v16bf v; } a;
            // 16-bit A layout: lane<16 holds K0-7 & K16-23; lane>=16 holds K8-15 & K24-31
            const uint4* pa = (const uint4*)&lA[buf][wave * 16 + r][kk * 32 + hi * 8];
            a.u[0] = pa[0];
            a.u[1] = pa[2];      // +16 halfwords
            #pragma unroll
            for (int j = 0; j < 8; j++) {
                union { uint4 u[2]; v16bf v; } b;
                const uint4* pb = (const uint4*)&lB[buf][j * 16 + r][kk * 32 + hi * 8];
                b.u[0] = pb[0];
                b.u[1] = pb[2];
                acc[j] = __builtin_amdgcn_wmma_f32_16x16x32_bf16(
                    false, a.v, false, b.v, (short)0, acc[j], false, false);
            }
        }

        // ---- grid-wide ADC max of this tile's partial sums ----
        float m = 0.f;
        #pragma unroll
        for (int j = 0; j < 8; j++)
            #pragma unroll
            for (int e = 0; e < 8; e++) m = fmaxf(m, fabsf(acc[j][e]));
        for (int s = 16; s; s >>= 1) m = fmaxf(m, __shfl_xor(m, s, 32));
        if (lane == 0) sMax[wave] = m;
        __syncthreads();
        if (tid == 0) {
            float bmax = sMax[0];
            for (int i = 1; i < 8; i++) bmax = fmaxf(bmax, sMax[i]);
            atomicMax(&ctl[32 + t], __float_as_uint(bmax));
            __threadfence();
            atomicAdd(&ctl[64 + t], 1u);
            while (__hip_atomic_load(&ctl[64 + t], __ATOMIC_ACQUIRE,
                                     __HIP_MEMORY_SCOPE_AGENT) < (unsigned)NBLK)
                __builtin_amdgcn_s_sleep(1);
        }
        __syncthreads();

        float mx = __uint_as_float(__hip_atomic_load(&ctl[32 + t], __ATOMIC_RELAXED,
                                                     __HIP_MEMORY_SCOPE_AGENT));
        float step = mx > 0.f ? mx * (1.f / 127.f) : 1.f;
        float inv  = 1.f / step;
        #pragma unroll
        for (int j = 0; j < 8; j++) {
            #pragma unroll
            for (int e = 0; e < 8; e++) {
                float q = fminf(fmaxf(rintf(acc[j][e] * inv), -127.f), 127.f);
                tot[j][e] += q * step;
            }
            acc[j] = vzero;
        }
    }

    // ---- epilogue: decode scale + bias, write fp32 ----
    #pragma unroll
    for (int j = 0; j < 8; j++) {
        int n = bn * BN + j * 16 + r;
        float bv = bias[n];
        #pragma unroll
        for (int e = 0; e < 8; e++) {
            int mrow = bm * BM + wave * 16 + e + hi * 8;   // C layout: lane>=16 -> M+8
            out[(size_t)mrow * OUT_DIM + n] = tot[j][e] * OUT_SCALE + bv;
        }
    }
}

extern "C" void kernel_launch(void* const* d_in, const int* in_sizes, int n_in,
                              void* d_out, int out_size, void* d_ws, size_t ws_size,
                              hipStream_t stream) {
    const float* x      = (const float*)d_in[0];
    const float* W      = (const float*)d_in[1];
    const float* bias   = (const float*)d_in[2];
    const float* d2dp   = (const float*)d_in[3];
    const float* d2dn   = (const float*)d_in[4];
    const float* noisep = (const float*)d_in[5];
    const float* noisen = (const float*)d_in[6];

    unsigned*       ctl = (unsigned*)d_ws;
    unsigned short* xq  = (unsigned short*)((char*)d_ws + XQ_OFF);
    unsigned short* g   = (unsigned short*)((char*)d_ws + G_OFF);
    float*          out = (float*)d_out;

    mx_init_kernel<<<1, 256, 0, stream>>>(ctl);
    mx_absmax_kernel<<<1024, 256, 0, stream>>>(x, ctl, B_DIM * IN_DIM / 4);
    mx_quantx_kernel<<<1024, 256, 0, stream>>>(x, ctl, xq, B_DIM * IN_DIM / 4);
    mx_geff_kernel<<<2048, 256, 0, stream>>>(W, d2dp, d2dn, noisep, noisen, g,
                                             OUT_DIM * IN_DIM / 4);
    mx_gemm_kernel<<<NBLK, 256, 0, stream>>>(xq, g, bias, ctl, out);
}
